// QuantizedMatmul_33389075759471
// MI455X (gfx1250) — matrix operand order repaired and checked by reading, not compiled, not run on hardware
//
#include <hip/hip_runtime.h>
#include <hip/hip_bf16.h>

typedef __attribute__((ext_vector_type(16))) _Float16 v16h;
typedef __attribute__((ext_vector_type(8)))  _Float16 v8h;
typedef __attribute__((ext_vector_type(8)))  float    v8f;

#define NBINS 1024
#define BATCH 8
#define MDIM 2048
#define KDIM 64
#define NDIM 2048
#define NX (2u*4u*2048u*64u)           // 1,048,576
#define NY (2u*4u*64u*2048u)           // 1,048,576
#define NO (2ull*4ull*2048ull*2048ull) // 33,554,432
#define EPSQ 1e-8f
#define SLM_CUTOFF 0.05f

// ---------- helpers ----------
__device__ __forceinline__ unsigned encF(float f) {
    unsigned u = __float_as_uint(f);
    return (u & 0x80000000u) ? ~u : (u | 0x80000000u);
}
__device__ __forceinline__ float decF(unsigned u) {
    unsigned v = (u & 0x80000000u) ? (u & 0x7FFFFFFFu) : ~u;
    return __uint_as_float(v);
}
// deterministic per-element uniform noise in [-0.5, 0.5)
__device__ __forceinline__ float unoise(unsigned idx) {
    unsigned h = idx * 0x9E3779B9u + 0x7F4A7C15u;
    h ^= h >> 16; h *= 0x85EBCA6Bu;
    h ^= h >> 13; h *= 0xC2B2AE35u;
    h ^= h >> 16;
    return (float)(h >> 8) * (1.0f / 16777216.0f) - 0.5f;
}

// ---------- stats init ----------
__global__ void qm_init_kernel(unsigned* __restrict__ mm, unsigned* __restrict__ hist) {
    int i = blockIdx.x * blockDim.x + threadIdx.x;
    if (i < 6) mm[i] = (i & 1) ? 0u : 0xFFFFFFFFu;   // even: min-enc, odd: max-enc
    if (i < 3 * NBINS) hist[i] = 0u;
}

// ---------- global min/max (block reduce + one atomic per block) ----------
__global__ __launch_bounds__(256) void qm_minmax_kernel(const float* __restrict__ p, size_t n,
                                                        unsigned* __restrict__ mm, int slot) {
    __shared__ float smn[256];
    __shared__ float smx[256];
    float lmn = INFINITY, lmx = -INFINITY;
    for (size_t i = (size_t)blockIdx.x * blockDim.x + threadIdx.x; i < n;
         i += (size_t)gridDim.x * blockDim.x) {
        float v = p[i];
        lmn = fminf(lmn, v);
        lmx = fmaxf(lmx, v);
    }
    int tid = threadIdx.x;
    smn[tid] = lmn; smx[tid] = lmx;
    __syncthreads();
    #pragma unroll
    for (int s = 128; s > 0; s >>= 1) {
        if (tid < s) {
            smn[tid] = fminf(smn[tid], smn[tid + s]);
            smx[tid] = fmaxf(smx[tid], smx[tid + s]);
        }
        __syncthreads();
    }
    if (tid == 0) {
        atomicMin(&mm[2 * slot],     encF(smn[0]));
        atomicMax(&mm[2 * slot + 1], encF(smx[0]));
    }
}

// ---------- histogram with LDS privatization ----------
__global__ __launch_bounds__(256) void qm_hist_kernel(const float* __restrict__ p, size_t n,
                                                      const unsigned* __restrict__ mm, int slot,
                                                      unsigned* __restrict__ hist) {
    __shared__ unsigned lh[NBINS];
    for (int i = threadIdx.x; i < NBINS; i += blockDim.x) lh[i] = 0u;
    __syncthreads();
    float mn = decF(mm[2 * slot]);
    float mx = decF(mm[2 * slot + 1]);
    float scale = (float)NBINS / fmaxf(mx - mn, 1e-12f);
    for (size_t i = (size_t)blockIdx.x * blockDim.x + threadIdx.x; i < n;
         i += (size_t)gridDim.x * blockDim.x) {
        int b = (int)((p[i] - mn) * scale);
        b = min(max(b, 0), NBINS - 1);
        atomicAdd(&lh[b], 1u);
    }
    __syncthreads();
    for (int i = threadIdx.x; i < NBINS; i += blockDim.x)
        if (lh[i]) atomicAdd(&hist[i], lh[i]);
}

// ---------- percentile resolve (QuantizeStats semantics) ----------
__global__ void qm_pct_kernel(const unsigned* __restrict__ mm, const unsigned* __restrict__ hist,
                              float* __restrict__ res, int slot, float pct, double n) {
    float mn = decF(mm[2 * slot]);
    float mx = decF(mm[2 * slot + 1]);
    double target = (double)pct / 100.0 * (n - 1.0);
    float binw = (mx - mn) / (float)NBINS;
    double cum = 0.0;
    float val = mx;
    for (int b = 0; b < NBINS; ++b) {
        double c = (double)hist[b];
        if (cum + c > target) {
            double frac = (c > 0.0) ? (target - cum) / c : 0.0;
            val = mn + ((float)b + (float)frac) * binw;
            break;
        }
        cum += c;
    }
    float pmx = val;
    float pmn = (mn >= 0.0f) ? mn : -pmx;   // min = t.min() if >=0 else -max
    res[2 * slot]     = pmn;
    res[2 * slot + 1] = pmx;
}

// ---------- deterministic fake-quant of x -> f16 (row-major, unchanged layout) ----------
__global__ __launch_bounds__(256) void qm_quantx_kernel(const float* __restrict__ src,
                                                        _Float16* __restrict__ dst,
                                                        const float* __restrict__ res, size_t n) {
    float mn = res[0], mx = res[1];
    float diff = mx - mn;
    float s   = 255.0f / (diff + EPSQ);
    float inv = diff * (1.0f / 255.0f);
    for (size_t i = (size_t)blockIdx.x * blockDim.x + threadIdx.x; i < n;
         i += (size_t)gridDim.x * blockDim.x) {
        float v = fminf(fmaxf(src[i], mn), mx);
        dst[i] = (_Float16)(rintf((v - mn) * s) * inv + mn);
    }
}

// ---------- fused fake-quant + SLM-LUT clip + transpose of y -> yqT[b][n][k] ----------
// LDS 64x64 tile (66-half pitch: 33-dword stride -> conflict-free bank rotation)
__global__ __launch_bounds__(256) void qm_quantTy_kernel(const float* __restrict__ y,
                                                         _Float16* __restrict__ yqT,
                                                         const float* __restrict__ res) {
    __shared__ _Float16 tile[64][66];
    const int bat = blockIdx.z;
    const int n0  = blockIdx.x * 64;
    float mn = res[2], mx = res[3];
    float diff = mx - mn;
    float s    = 255.0f / (diff + EPSQ);
    float inv  = diff * (1.0f / 255.0f);
    float lmsp = SLM_CUTOFF * mx;
    float lmsn = SLM_CUTOFF * fabsf(mn);
    const float* yb = y + (size_t)bat * KDIM * NDIM;
    #pragma unroll
    for (int i = 0; i < 16; ++i) {                 // coalesced reads along n
        int e = threadIdx.x + i * 256;
        int k = e >> 6, n = e & 63;
        float v = fminf(fmaxf(yb[(size_t)k * NDIM + n0 + n], mn), mx);
        float q = rintf((v - mn) * s) * inv + mn;
        float pos = fmaxf(q, 0.0f);
        float neg = q - pos;
        pos = (pos < lmsp) ? lmsp : pos;
        neg = (neg > -lmsn) ? -lmsn : neg;
        tile[k][n] = (_Float16)(pos + neg);
    }
    __syncthreads();
    _Float16* ob = yqT + ((size_t)bat * NDIM + n0) * KDIM;
    #pragma unroll
    for (int i = 0; i < 16; ++i) {                 // coalesced writes along k
        int e = threadIdx.x + i * 256;
        int n = e >> 6, k = e & 63;
        ob[(size_t)n * KDIM + k] = tile[k][n];
    }
}

// ---------- WMMA GEMM + fused output min/max epilogue ----------
// out[b] = xq[b](2048x64, row-major) * yqT[b](2048x64 N-major == 64x2048 K x N)
__global__ __launch_bounds__(128) void qm_gemm_kernel(const _Float16* __restrict__ xq,
                                                      const _Float16* __restrict__ yqT,
                                                      float* __restrict__ out,
                                                      unsigned* __restrict__ mm) {
    const int lane = threadIdx.x & 31;
    const int wave = threadIdx.x >> 5;
    const int bat  = blockIdx.z;
    const int n0   = blockIdx.x * 16;
    const int m0   = (blockIdx.y * 4 + wave) * 16;

    const _Float16* xb = xq + (size_t)bat * MDIM * KDIM;
    float*          ob = out + (size_t)bat * MDIM * NDIM;

    const int lh = lane & 15;       // lane within half-wave
    const int cs = lane >> 4;       // half-wave select

    // A fragments: row-major x, contiguous K order (k = 16*half + s) matching B
    const v8h* arp = (const v8h*)(xb + (size_t)(m0 + lh) * KDIM);
    v8h A0lo = arp[2 * cs],     A0hi = arp[2 * cs + 1];
    v8h A1lo = arp[4 + 2 * cs], A1hi = arp[5 + 2 * cs];
    v16h a0 = __builtin_shufflevector(A0lo, A0hi, 0,1,2,3,4,5,6,7,8,9,10,11,12,13,14,15);
    v16h a1 = __builtin_shufflevector(A1lo, A1hi, 0,1,2,3,4,5,6,7,8,9,10,11,12,13,14,15);

    // B fragments: yqT has K contiguous -> one 32B v16h load per fragment.
    // Lanes 0-15 hold low-K half, 16-31 high-K half, consecutive k per half-lane-group.
    const v16h* brow = (const v16h*)(yqT + (size_t)(bat * NDIM + n0 + lh) * KDIM);
    v16h b0 = brow[cs];         // K 0..31
    v16h b1 = brow[2 + cs];     // K 32..63

    v8f c = {};
    c = __builtin_amdgcn_wmma_f32_16x16x32_f16(false, a0, false, b0, (short)0, c, false, false);
    c = __builtin_amdgcn_wmma_f32_16x16x32_f16(false, a1, false, b1, (short)0, c, false, false);

    // store D per 16x16 f32 C/D layout
    const int nOut  = n0 + lh;
    const int mBase = m0 + cs * 8;
    #pragma unroll
    for (int r = 0; r < 8; ++r)
        ob[(size_t)(mBase + r) * NDIM + nOut] = c[r];

    // fused output min/max (saves a full 134MB read pass)
    float lmn = c[0], lmx = c[0];
    #pragma unroll
    for (int r = 1; r < 8; ++r) {
        lmn = fminf(lmn, c[r]);
        lmx = fmaxf(lmx, c[r]);
    }
    __shared__ float smn[128];
    __shared__ float smx[128];
    const int tid = threadIdx.x;
    smn[tid] = lmn; smx[tid] = lmx;
    __syncthreads();
    #pragma unroll
    for (int s = 64; s > 0; s >>= 1) {
        if (tid < s) {
            smn[tid] = fminf(smn[tid], smn[tid + s]);
            smx[tid] = fmaxf(smx[tid], smx[tid + s]);
        }
        __syncthreads();
    }
    if (tid == 0) {
        atomicMin(&mm[4], encF(smn[0]));
        atomicMax(&mm[5], encF(smx[0]));
    }
}

// ---------- stochastic-rounded fake quant of output (in place) ----------
__global__ __launch_bounds__(256) void qm_squant_kernel(float* __restrict__ o,
                                                        const float* __restrict__ res, size_t n) {
    float mn = res[4], mx = res[5];
    float diff = mx - mn;
    float s   = 255.0f / (diff + EPSQ);
    float inv = diff * (1.0f / 255.0f);
    for (size_t i = (size_t)blockIdx.x * blockDim.x + threadIdx.x; i < n;
         i += (size_t)gridDim.x * blockDim.x) {
        float v = fminf(fmaxf(o[i], mn), mx);
        float q = rintf((v - mn) * s + unoise((unsigned)i)) * inv + mn;
        o[i] = q;
    }
}

extern "C" void kernel_launch(void* const* d_in, const int* in_sizes, int n_in,
                              void* d_out, int out_size, void* d_ws, size_t ws_size,
                              hipStream_t stream) {
    const float* x = (const float*)d_in[0];
    const float* y = (const float*)d_in[1];
    float* out = (float*)d_out;

    char* ws = (char*)d_ws;
    unsigned*  mm   = (unsigned*)(ws + 0);      // 6 encoded min/max
    float*     res  = (float*)(ws + 64);        // 3 resolved (mn,mx) pairs
    unsigned*  hist = (unsigned*)(ws + 128);    // 3 * NBINS
    _Float16*  xq   = (_Float16*)(ws + 16384);
    _Float16*  yqT  = (_Float16*)(ws + 16384 + (size_t)NX * sizeof(_Float16));

    // stage 0: reset stats state (ws is not re-poisoned between replays)
    qm_init_kernel<<<(3 * NBINS + 255) / 256, 256, 0, stream>>>(mm, hist);

    // stage 1: input calibration (min/max -> histogram -> percentile)
    qm_minmax_kernel<<<256, 256, 0, stream>>>(x, (size_t)NX, mm, 0);
    qm_minmax_kernel<<<256, 256, 0, stream>>>(y, (size_t)NY, mm, 1);
    qm_hist_kernel<<<256, 256, 0, stream>>>(x, (size_t)NX, mm, 0, hist);
    qm_hist_kernel<<<256, 256, 0, stream>>>(y, (size_t)NY, mm, 1, hist + NBINS);
    qm_pct_kernel<<<1, 1, 0, stream>>>(mm, hist,         res, 0, 99.99f, (double)NX);
    qm_pct_kernel<<<1, 1, 0, stream>>>(mm, hist + NBINS, res, 1, 98.0f,  (double)NY);

    // stage 2: fake-quant inputs to f16 (y fused with LDS-tiled transpose -> K-contiguous)
    qm_quantx_kernel<<<512, 256, 0, stream>>>(x, xq, res, (size_t)NX);
    dim3 tGrid(NDIM / 64, 1, BATCH);
    qm_quantTy_kernel<<<tGrid, 256, 0, stream>>>(y, yqT, res);

    // stage 3: batched WMMA GEMM (2x v_wmma_f32_16x16x32_f16) + fused out min/max
    dim3 gGrid(NDIM / 16, MDIM / 64, BATCH);
    qm_gemm_kernel<<<gGrid, 128, 0, stream>>>(xq, yqT, out, mm);

    // stage 4: output histogram -> percentile -> stochastic fake quant in place
    qm_hist_kernel<<<1024, 256, 0, stream>>>(out, (size_t)NO, mm, 2, hist + 2 * NBINS);
    qm_pct_kernel<<<1, 1, 0, stream>>>(mm, hist + 2 * NBINS, res, 2, 99.9999f, (double)NO);
    qm_squant_kernel<<<1024, 256, 0, stream>>>(out, res, (size_t)NO);
}
